// SelfAttention_sig_7438883357605
// MI455X (gfx1250) — matrix operand order repaired and checked by reading, not compiled, not run on hardware
//
#include <hip/hip_runtime.h>

typedef __attribute__((ext_vector_type(16))) __bf16 v16bf;
typedef __attribute__((ext_vector_type(8)))  float  v8f;
typedef __attribute__((ext_vector_type(4)))  float  v4f;
typedef __attribute__((ext_vector_type(4)))  unsigned int v4u;
typedef __attribute__((ext_vector_type(2)))  unsigned int v2u;

union ABu { v16bf v; unsigned short s[16]; v4u q[2]; };
union U8  { v4u q; unsigned short s[8]; };

__device__ __forceinline__ unsigned short f2bf(float f) {
  return __builtin_bit_cast(unsigned short, (__bf16)f);   // hw cvt
}

__device__ __forceinline__ v8f wmma_bf16(v16bf a, v16bf b, v8f c) {
  // D = A(16x32 bf16) * B(32x16 bf16) + C(16x16 f32)
  return __builtin_amdgcn_wmma_f32_16x16x32_bf16(false, a, false, b, (short)0, c, false, false);
}

// ---------------- weight convert + transpose: Wt[n][k] bf16 ----------------
__global__ void wtcvt_kernel(const float* __restrict__ w, unsigned short* __restrict__ wt) {
  int t = blockIdx.x * 256 + threadIdx.x;     // 512*512 elems
  int k = t >> 9, n = t & 511;
  wt[n * 512 + k] = f2bf(w[t]);
}

// ---------------- activation convert fp32 -> bf16 (4 elems/thread) ----------------
__global__ void xcvt_kernel(const float* __restrict__ x, unsigned short* __restrict__ o) {
  int t4 = (blockIdx.x * 256 + threadIdx.x) * 4;
  v4f x4 = *(const v4f*)(x + t4);
  unsigned u01 = (unsigned)f2bf(x4[0]) | ((unsigned)f2bf(x4[1]) << 16);
  unsigned u23 = (unsigned)f2bf(x4[2]) | ((unsigned)f2bf(x4[3]) << 16);
  v2u pk = {u01, u23};
  *(v2u*)(o + t4) = pk;
}

// ------- 5 fused projections: Xb[16384,512]bf16 @ Wt -> heads -------
// Each wave owns a 16x64 strip: A fragment loaded once per k-step, 4 WMMAs.
__global__ void proj_kernel(const unsigned short* __restrict__ XQb, const unsigned short* __restrict__ XKb,
                            const unsigned short* __restrict__ XVb, const unsigned short* __restrict__ WT,
                            const float* __restrict__ bq, const float* __restrict__ bk,
                            const float* __restrict__ bv, const float* __restrict__ bq1,
                            const float* __restrict__ bk1,
                            unsigned short* __restrict__ QH, unsigned short* __restrict__ KH,
                            unsigned short* __restrict__ Q1H, unsigned short* __restrict__ K1H,
                            unsigned short* __restrict__ VT) {
  const int p    = blockIdx.z;                     // 0:Q 1:K 2:V 3:Q1 4:K1
  const int lane = threadIdx.x & 31, wid = threadIdx.x >> 5;
  const int half = lane >> 4, l15 = lane & 15;
  const int mbase  = blockIdx.y * 16;              // row tile of 16384
  const int nbase0 = blockIdx.x * 256 + wid * 64;  // 64-wide col strip of 512

  const unsigned short* Xb  = (p == 0 || p == 3) ? XQb : (p == 1 || p == 4) ? XKb : XVb;
  const float* bias = (p == 0) ? bq : (p == 1) ? bk : (p == 2) ? bv : (p == 3) ? bq1 : bk1;
  const unsigned short* Wt = WT + (size_t)p * 262144;

  const unsigned short* arow = Xb + (size_t)(mbase + l15) * 512;
  const unsigned short* brow0 = Wt + (size_t)(nbase0 + l15) * 512;   // tiles at +t*16*512

  v8f c[4] = {{}, {}, {}, {}};
  for (int kc = 0; kc < 512; kc += 32) {
    ABu a;
    a.q[0] = *(const v4u*)(arow + kc + half * 8);
    a.q[1] = *(const v4u*)(arow + kc + 16 + half * 8);
#pragma unroll
    for (int t = 0; t < 4; ++t) {
      const v4u* bp = (const v4u*)(brow0 + (size_t)t * 16 * 512 + kc + half * 8);
      ABu b; b.q[0] = bp[0]; b.q[1] = bp[2];
      c[t] = wmma_bf16(a.v, b.v, c[t]);
    }
    if (kc + 64 < 512) __builtin_prefetch(arow + kc + 64, 0, 1);
  }

  const int bb = mbase >> 10, srow = mbase & 1023;
#pragma unroll
  for (int t = 0; t < 4; ++t) {
    const int n = nbase0 + t * 16 + l15, h = n >> 6, d = n & 63;
    const float bv_ = bias[n];
    if (p == 2) {
      // V stored transposed: VT[bh][d][s]; C rows are 8 consecutive s -> one b128 store
      U8 u;
#pragma unroll
      for (int r = 0; r < 8; ++r) u.s[r] = f2bf(c[t][r] + bv_);
      size_t idx = ((size_t)(bb * 8 + h) * 64 + d) * 1024 + srow + half * 8;
      *(v4u*)(VT + idx) = u.q;
    } else {
      unsigned short* o = (p == 0 ? QH : p == 1 ? KH : p == 3 ? Q1H : K1H)
                          + ((size_t)(bb * 8 + h) * 1024 + srow) * 64 + d;
#pragma unroll
      for (int r = 0; r < 8; ++r) o[(size_t)(r + half * 8) * 64] = f2bf(c[t][r] + bv_);
    }
  }
}

// ---------------- attention: block owns (b, 16 query rows), loops all heads ----------------
__global__ void attn_kernel(const unsigned short* __restrict__ QH, const unsigned short* __restrict__ KH,
                            const unsigned short* __restrict__ Q1H, const unsigned short* __restrict__ K1H,
                            const unsigned short* __restrict__ VT, unsigned short* __restrict__ XC,
                            const int* __restrict__ mask, float* __restrict__ ratt) {
  __shared__ float e_s[16 * 1024];          // 64KB energy -> exp(p)
  __shared__ float sg [16 * 1024];          // 64KB sigmoid(sample)
  __shared__ unsigned short abf[16 * 1024]; // 32KB attention probs (bf16, WMMA A operand)
  const int lane = threadIdx.x & 31, wid = threadIdx.x >> 5;   // 4 waves
  const int half = lane >> 4, l15 = lane & 15;
  const int b = blockIdx.y, qbase = blockIdx.x * 16;

  for (int h = 0; h < 8; ++h) {
    const size_t bh = (size_t)(b * 8 + h) * 65536;
    ABu qf[2], q1f[2];
#pragma unroll
    for (int j = 0; j < 2; ++j) {
      const unsigned short* qr  = QH  + bh + (size_t)(qbase + l15) * 64 + j * 32;
      qf[j].q[0]  = *(const v4u*)(qr + half * 8);
      qf[j].q[1]  = *(const v4u*)(qr + 16 + half * 8);
      const unsigned short* q1r = Q1H + bh + (size_t)(qbase + l15) * 64 + j * 32;
      q1f[j].q[0] = *(const v4u*)(q1r + half * 8);
      q1f[j].q[1] = *(const v4u*)(q1r + 16 + half * 8);
    }
    // energy + sample tiles across all 1024 keys (16 col tiles per wave)
    for (int ct = wid; ct < 64; ct += 4) {
      const int col0 = ct * 16;
      v8f ec = {}, sc = {};
#pragma unroll
      for (int j = 0; j < 2; ++j) {
        const unsigned short* kr  = KH  + bh + (size_t)(col0 + l15) * 64 + j * 32 + half * 8;
        ABu kf;  kf.q[0]  = ((const v4u*)kr)[0];  kf.q[1]  = ((const v4u*)kr)[2];
        ec = wmma_bf16(qf[j].v, kf.v, ec);
        const unsigned short* k1r = K1H + bh + (size_t)(col0 + l15) * 64 + j * 32 + half * 8;
        ABu k1f; k1f.q[0] = ((const v4u*)k1r)[0]; k1f.q[1] = ((const v4u*)k1r)[2];
        sc = wmma_bf16(q1f[j].v, k1f.v, sc);
      }
      const int col = col0 + l15;
      const int mk = mask[b * 1024 + col];
#pragma unroll
      for (int r = 0; r < 8; ++r) {
        const int row = r + half * 8;
        e_s[row * 1024 + col] = mk ? ec[r] * 0.125f : -__builtin_inff();
        sg [row * 1024 + col] = 1.0f / (1.0f + __expf(-sc[r] * 0.125f));
      }
    }
    __syncthreads();
    // softmax * sigmoid -> abf (bf16); accumulate r_att (block owns rows, L2-resident RMW)
    for (int rr = 0; rr < 4; ++rr) {
      const int row = wid * 4 + rr;
      float* er = e_s + row * 1024;
      const float* sgr = sg + row * 1024;
      float mx = -__builtin_inff();
      for (int cc = lane; cc < 1024; cc += 32) mx = fmaxf(mx, er[cc]);
      for (int o = 16; o > 0; o >>= 1) mx = fmaxf(mx, __shfl_xor(mx, o, 32));
      float sum = 0.f;
      for (int cc = lane; cc < 1024; cc += 32) { float pv = __expf(er[cc] - mx); er[cc] = pv; sum += pv; }
      for (int o = 16; o > 0; o >>= 1) sum += __shfl_xor(sum, o, 32);
      const float inv = 1.0f / sum;
      float* rp = ratt + (size_t)(b * 1024 + qbase + row) * 1024;
      for (int cc = lane; cc < 1024; cc += 32) {
        float att = er[cc] * inv * sgr[cc];
        abf[row * 1024 + cc] = f2bf(att);
        if (h == 0) rp[cc] = att * 0.125f; else rp[cc] += att * 0.125f;   // mean over 8 heads
      }
    }
    __syncthreads();
    // context: attention[16,1024](bf16 in LDS) @ V[1024,64]; wave owns one 16-wide d tile
    v8f cacc = {};
    const unsigned short* vtb = VT + bh + (size_t)(wid * 16 + l15) * 1024;
    for (int kc = 0; kc < 32; ++kc) {
      const int cb = kc * 32;
      const unsigned short* ar = abf + l15 * 1024 + cb + half * 8;
      ABu af; af.q[0] = *(const v4u*)ar; af.q[1] = *(const v4u*)(ar + 16);
      const v4u* vp = (const v4u*)(vtb + cb + half * 8);
      ABu bf; bf.q[0] = vp[0]; bf.q[1] = vp[2];
      cacc = wmma_bf16(af.v, bf.v, cacc);
      if (kc < 31) __builtin_prefetch(vtb + cb + 64, 0, 1);
    }
    const int d = h * 64 + wid * 16 + l15;
#pragma unroll
    for (int r = 0; r < 8; ++r) {
      const int s = qbase + r + half * 8;
      XC[(size_t)(b * 1024 + s) * 512 + d] = f2bf(cacc[r]);
    }
    __syncthreads();
  }
}

// ------- final FC: XC[16384,512] @ wfc + bfc -> fp32 out (16x64 strip per wave) -------
__global__ void fc_kernel(const unsigned short* __restrict__ XC, const unsigned short* __restrict__ WtFC,
                          const float* __restrict__ bias, float* __restrict__ out) {
  const int lane = threadIdx.x & 31, wid = threadIdx.x >> 5;
  const int half = lane >> 4, l15 = lane & 15;
  const int mbase  = blockIdx.y * 16;
  const int nbase0 = blockIdx.x * 256 + wid * 64;
  const unsigned short* arow  = XC   + (size_t)(mbase + l15) * 512;
  const unsigned short* brow0 = WtFC + (size_t)(nbase0 + l15) * 512;
  v8f c[4] = {{}, {}, {}, {}};
  for (int kc = 0; kc < 512; kc += 32) {
    ABu a;
    a.q[0] = *(const v4u*)(arow + kc + half * 8);
    a.q[1] = *(const v4u*)(arow + kc + 16 + half * 8);
#pragma unroll
    for (int t = 0; t < 4; ++t) {
      const v4u* bp = (const v4u*)(brow0 + (size_t)t * 16 * 512 + kc + half * 8);
      ABu b2; b2.q[0] = bp[0]; b2.q[1] = bp[2];
      c[t] = wmma_bf16(a.v, b2.v, c[t]);
    }
    if (kc + 64 < 512) __builtin_prefetch(arow + kc + 64, 0, 1);
  }
#pragma unroll
  for (int t = 0; t < 4; ++t) {
    const int n = nbase0 + t * 16 + l15;
    const float bv_ = bias[n];
#pragma unroll
    for (int r = 0; r < 8; ++r)
      out[(size_t)(mbase + r + half * 8) * 512 + n] = c[t][r] + bv_;
  }
}

extern "C" void kernel_launch(void* const* d_in, const int* in_sizes, int n_in,
                              void* d_out, int out_size, void* d_ws, size_t ws_size,
                              hipStream_t stream) {
  (void)in_sizes; (void)n_in; (void)out_size; (void)ws_size;
  const float* q_in = (const float*)d_in[0];
  const float* k_in = (const float*)d_in[1];
  const float* v_in = (const float*)d_in[2];
  const int*   mask = (const int*)d_in[3];
  const float* wq  = (const float*)d_in[4];  const float* bq  = (const float*)d_in[5];
  const float* wk  = (const float*)d_in[6];  const float* bk  = (const float*)d_in[7];
  const float* wv  = (const float*)d_in[8];  const float* bv  = (const float*)d_in[9];
  const float* wq1 = (const float*)d_in[10]; const float* bq1 = (const float*)d_in[11];
  const float* wk1 = (const float*)d_in[12]; const float* bk1 = (const float*)d_in[13];
  const float* wfc = (const float*)d_in[14]; const float* bfc = (const float*)d_in[15];

  // workspace (ushort elems): 6 transposed bf16 weights, bf16 activations (q/k/v),
  // Q/K/Q1/K1 heads, transposed V.  XC aliases XQb (dead after proj).  ~131 MB.
  unsigned short* WT = (unsigned short*)d_ws;
  size_t off = 6u * 262144u;
  unsigned short* XQb = WT + off;  off += 8388608;
  unsigned short* XKb = WT + off;  off += 8388608;
  unsigned short* XVb = WT + off;  off += 8388608;
  unsigned short* QH  = WT + off;  off += 8388608;
  unsigned short* KH  = WT + off;  off += 8388608;
  unsigned short* Q1H = WT + off;  off += 8388608;
  unsigned short* K1H = WT + off;  off += 8388608;
  unsigned short* VTp = WT + off;  off += 8388608;
  unsigned short* XC  = XQb;       // reuse: query activations dead after proj

  float* out  = (float*)d_out;
  float* ratt = out + 8388608;     // [B,S,S] after x [B,S,D]

  wtcvt_kernel<<<1024, 256, 0, stream>>>(wq,  WT + 0 * 262144);
  wtcvt_kernel<<<1024, 256, 0, stream>>>(wk,  WT + 1 * 262144);
  wtcvt_kernel<<<1024, 256, 0, stream>>>(wv,  WT + 2 * 262144);
  wtcvt_kernel<<<1024, 256, 0, stream>>>(wq1, WT + 3 * 262144);
  wtcvt_kernel<<<1024, 256, 0, stream>>>(wk1, WT + 4 * 262144);
  wtcvt_kernel<<<1024, 256, 0, stream>>>(wfc, WT + 5 * 262144);

  xcvt_kernel<<<8192, 256, 0, stream>>>(q_in, XQb);
  xcvt_kernel<<<8192, 256, 0, stream>>>(k_in, XKb);
  xcvt_kernel<<<8192, 256, 0, stream>>>(v_in, XVb);

  proj_kernel<<<dim3(2, 1024, 5), 128, 0, stream>>>(XQb, XKb, XVb, WT,
      bq, bk, bv, bq1, bk1, QH, KH, Q1H, K1H, VTp);

  attn_kernel<<<dim3(64, 16), 128, 0, stream>>>(QH, KH, Q1H, K1H, VTp, XC, mask, ratt);

  fc_kernel<<<dim3(2, 1024), 128, 0, stream>>>(XC, WT + 5 * 262144, bfc, out);
}
